// MultimodalAttentionGNN_50259707298591
// MI455X (gfx1250) — matrix operand order repaired and checked by reading, not compiled, not run on hardware
//
#include <hip/hip_runtime.h>
#include <hip/hip_bf16.h>

typedef __attribute__((ext_vector_type(16))) _Float16 v16h;
typedef __attribute__((ext_vector_type(8)))  float    v8f;

union AFrag { v16h v; _Float16 h[16]; };
union BFrag { v16h v; uint4 q[2]; };

__device__ __forceinline__ float warp_reduce_sum(float x) {
#pragma unroll
    for (int o = 16; o > 0; o >>= 1) x += __shfl_xor(x, o, 32);
    return x;
}

// ---------------------------------------------------------------------------
// f32 -> f16 weight pre-conversion (run once per weight matrix)
// ---------------------------------------------------------------------------
__global__ void cvt_f32_f16(const float* __restrict__ src, _Float16* __restrict__ dst,
                            int count)
{
    int t = blockIdx.x * blockDim.x + threadIdx.x;
    if (t < count) dst[t] = (_Float16)src[t];
}

// ---------------------------------------------------------------------------
// Generic WMMA GEMM:  Y[M,N] = (gather(X, idx)%mod) @ W16[K,N] + bias + resid
// Block: 256 threads = 8 waves. Each wave -> 16 rows x 128 cols (8 WMMA tiles),
// so one A fragment feeds 8 WMMAs. Grid: (ceil(M/128), N/128).
// A: f32->f16 on the fly (v_cvt_pk_f16_f32). B: f16 tile (32x128 = 8KB) brought
// into LDS with GLOBAL_LOAD_ASYNC_TO_LDS_B128 (2 x 16B per thread), completed
// with s_wait_asynccnt + workgroup barrier.
// ---------------------------------------------------------------------------
__global__ __launch_bounds__(256)
void gemm_wmma_f16(const float* __restrict__ X, int ldX,
                   const int* __restrict__ idx, int idxMod,
                   const _Float16* __restrict__ W16, int K, int N,
                   const float* __restrict__ bias,
                   const float* __restrict__ resid, int ldR,
                   float* __restrict__ Y, int ldY, int M)
{
    __shared__ __align__(16) _Float16 ldsB[32][128];  // 8 KB: 32 K-rows x 128 N-cols
    const int tid  = threadIdx.x;
    const int wid  = tid >> 5;
    const int lane = tid & 31;
    const int mBase = blockIdx.x * 128 + wid * 16;
    const int nBase = blockIdx.y * 128;

    // A row for this lane (16-bit A layout: lanes 0-15 / 16-31 both map M=lane&15)
    const int arow_m = mBase + (lane & 15);
    int srcRow = (arow_m < M) ? arow_m : 0;
    if (idx && arow_m < M) {
        srcRow = idx[arow_m];
        if (idxMod > 0) srcRow %= idxMod;
    }
    const float* arow = X + (size_t)srcRow * ldX;
    const int o1 = (lane >> 4) * 8;      // K sub-offset: 0 (lanes 0-15) or 8 (lanes 16-31)

    // B-tile cooperative assignment: 32 bytes (16 halves) per thread
    const int br  = tid >> 3;            // 0..31  : K row within tile
    const int bc0 = (tid & 7) * 16;      // 0..112 : N col
    const _Float16* wptr0 = W16 + (size_t)br * N + nBase + bc0;
    const unsigned myLds = (unsigned)(size_t)&ldsB[br][bc0];   // LDS offset (low bits)

    v8f acc[8] = {};
    const int Kfull = K & ~31;
    int kb = 0;

    // ---------------- main loop: no bounds checks ----------------
    for (; kb < Kfull; kb += 32) {
        // async B-tile -> LDS (2 x 16 bytes per thread, ASYNCcnt tracked)
        {
            const _Float16* gsrc = wptr0 + (size_t)kb * N;
            asm volatile("global_load_async_to_lds_b128 %0, %1, off"
                         :: "v"(myLds), "v"(gsrc) : "memory");
            asm volatile("global_load_async_to_lds_b128 %0, %1, off offset:16"
                         :: "v"(myLds), "v"(gsrc) : "memory");
        }

        // A fragment: 16 f16 per lane (K chunks o1..o1+7, o1+16..o1+23)
        AFrag A;
        {
            const float* ap = arow + kb + o1;
            float4 a0 = *(const float4*)(ap);
            float4 a1 = *(const float4*)(ap + 4);
            float4 a2 = *(const float4*)(ap + 16);
            float4 a3 = *(const float4*)(ap + 20);
            A.h[0]  = (_Float16)a0.x; A.h[1]  = (_Float16)a0.y;
            A.h[2]  = (_Float16)a0.z; A.h[3]  = (_Float16)a0.w;
            A.h[4]  = (_Float16)a1.x; A.h[5]  = (_Float16)a1.y;
            A.h[6]  = (_Float16)a1.z; A.h[7]  = (_Float16)a1.w;
            A.h[8]  = (_Float16)a2.x; A.h[9]  = (_Float16)a2.y;
            A.h[10] = (_Float16)a2.z; A.h[11] = (_Float16)a2.w;
            A.h[12] = (_Float16)a3.x; A.h[13] = (_Float16)a3.y;
            A.h[14] = (_Float16)a3.z; A.h[15] = (_Float16)a3.w;
        }

        asm volatile("s_wait_asynccnt 0x0" ::: "memory");
        __syncthreads();

#pragma unroll
        for (int ns = 0; ns < 8; ++ns) {
            BFrag Bf;
#pragma unroll
            for (int s = 0; s < 16; ++s)
                Bf.v[s] = ldsB[(s < 8) ? (o1 + s) : (16 + o1 + s - 8)][ns * 16 + (lane & 15)];
            acc[ns] = __builtin_amdgcn_wmma_f32_16x16x32_f16(
                false, A.v, false, Bf.v, (short)0, acc[ns], false, false);
        }
        __syncthreads();
    }

    // ---------------- tail step (K % 32 != 0, e.g. K = 1000) ----------------
    if (kb < K) {
#pragma unroll
        for (int j = 0; j < 16; ++j) {
            ldsB[br][bc0 + j] = (kb + br < K)
                ? W16[(size_t)(kb + br) * N + nBase + bc0 + j] : (_Float16)0.f;
        }
        __syncthreads();

        AFrag A;
        const int k0 = kb + o1;
#pragma unroll
        for (int j = 0; j < 8; ++j) {
            int ka = k0 + j, kc = k0 + 16 + j;
            A.h[j]     = (_Float16)((ka < K) ? arow[ka] : 0.f);
            A.h[8 + j] = (_Float16)((kc < K) ? arow[kc] : 0.f);
        }

#pragma unroll
        for (int ns = 0; ns < 8; ++ns) {
            BFrag Bf;
#pragma unroll
            for (int s = 0; s < 16; ++s)
                Bf.v[s] = ldsB[(s < 8) ? (o1 + s) : (16 + o1 + s - 8)][ns * 16 + (lane & 15)];
            acc[ns] = __builtin_amdgcn_wmma_f32_16x16x32_f16(
                false, A.v, false, Bf.v, (short)0, acc[ns], false, false);
        }
    }

    // ---- epilogue: C layout = VGPR r -> M=r (lanes 0-15) / M=8+r (lanes 16-31) ----
    const int colLane = lane & 15;
    const int mOff    = (lane < 16) ? 0 : 8;
#pragma unroll
    for (int ns = 0; ns < 8; ++ns) {
        const int col = nBase + ns * 16 + colLane;
        const float bv = bias ? bias[col] : 0.f;
#pragma unroll
        for (int r = 0; r < 8; ++r) {
            const int row = mBase + mOff + r;
            if (row < M) {
                float v = acc[ns][r] + bv;
                if (resid) v += resid[(size_t)row * ldR + col];
                Y[(size_t)row * ldY + col] = v;
            }
        }
    }
}

// ---------------------------------------------------------------------------
// x[:, 0:256] = emb_table[ids]   (first B rows user, next B rows item), ldY=512
// ---------------------------------------------------------------------------
__global__ void embed_copy(const float* __restrict__ uet, const float* __restrict__ iet,
                           const int* __restrict__ uid, const int* __restrict__ iid,
                           float* __restrict__ x, int B)
{
    int t = blockIdx.x * blockDim.x + threadIdx.x;   // over 2B*64 float4s
    int row = t >> 6;
    int c   = (t & 63) * 4;
    if (row >= 2 * B) return;
    const float* src = (row < B) ? (uet + (size_t)uid[row] * 256 + c)
                                 : (iet + (size_t)iid[row - B] * 256 + c);
    *(float4*)(x + (size_t)row * 512 + c) = *(const float4*)src;
}

// ---------------------------------------------------------------------------
// es[n,h] = <h[n,h,:], a_s[h,:]>, ed likewise. One wave per node.
// ---------------------------------------------------------------------------
__global__ __launch_bounds__(256)
void gat_scores(const float* __restrict__ h, const float* __restrict__ a_s,
                const float* __restrict__ a_d, float* __restrict__ es,
                float* __restrict__ ed, int n)
{
    int node = blockIdx.x * 8 + (threadIdx.x >> 5);
    int lane = threadIdx.x & 31;
    if (node >= n) return;
    const float* hr = h + (size_t)node * 256;
    float ps[4], pd[4];
#pragma unroll
    for (int hh = 0; hh < 4; ++hh) {
        int c0 = hh * 64 + lane;
        float v0 = hr[c0], v1 = hr[c0 + 32];
        ps[hh] = v0 * a_s[c0] + v1 * a_s[c0 + 32];
        pd[hh] = v0 * a_d[c0] + v1 * a_d[c0 + 32];
    }
#pragma unroll
    for (int hh = 0; hh < 4; ++hh) {
        ps[hh] = warp_reduce_sum(ps[hh]);
        pd[hh] = warp_reduce_sum(pd[hh]);
    }
    if (lane == 0) {
#pragma unroll
        for (int hh = 0; hh < 4; ++hh) {
            es[node * 4 + hh] = ps[hh];
            ed[node * 4 + hh] = pd[hh];
        }
    }
}

__global__ void init_layer(float* __restrict__ m, float* __restrict__ den,
                           float* __restrict__ agg, int n)
{
    size_t t = (size_t)blockIdx.x * blockDim.x + threadIdx.x;
    if (t < (size_t)n * 4) { m[t] = -__builtin_inff(); den[t] = 0.f; }
    if (t < (size_t)n * 256) agg[t] = 0.f;
}

__device__ __forceinline__ void atomicMaxF(float* addr, float val) {
    // order-preserving trick, valid with init = -inf
    if (val >= 0.f) atomicMax((int*)addr, __float_as_int(val));
    else            atomicMin((unsigned int*)addr, (unsigned int)__float_as_int(val));
}

__global__ void edge_max(const int* __restrict__ src, const int* __restrict__ dst,
                         const float* __restrict__ es, const float* __restrict__ ed,
                         float* __restrict__ ebuf, float* __restrict__ m, int E)
{
    int t = blockIdx.x * blockDim.x + threadIdx.x;
    if (t >= E * 4) return;
    int e = t >> 2, hh = t & 3;
    int s = src[e], d = dst[e];
    float v = es[s * 4 + hh] + ed[d * 4 + hh];
    v = (v > 0.f) ? v : 0.2f * v;                    // leaky_relu(0.2)
    ebuf[t] = v;
    atomicMaxF(&m[d * 4 + hh], v);
}

__global__ void fix_m(float* __restrict__ m, int n4)
{
    int t = blockIdx.x * blockDim.x + threadIdx.x;
    if (t < n4) { if (m[t] < -3.0e38f) m[t] = 0.f; }  // -inf (empty segment) -> 0
}

__global__ void edge_exp(const int* __restrict__ dst, float* __restrict__ ebuf,
                         const float* __restrict__ m, float* __restrict__ den, int E)
{
    int t = blockIdx.x * blockDim.x + threadIdx.x;
    if (t >= E * 4) return;
    int e = t >> 2, hh = t & 3;
    int d = dst[e];
    float ex = __expf(ebuf[t] - m[d * 4 + hh]);
    ebuf[t] = ex;
    atomicAdd(&den[d * 4 + hh], ex);
}

// wave per edge: agg[dst] += alpha * h[src]  (256 floats)
__global__ __launch_bounds__(256)
void edge_agg(const int* __restrict__ src, const int* __restrict__ dst,
              const float* __restrict__ ebuf, const float* __restrict__ den,
              const float* __restrict__ h, float* __restrict__ agg, int E)
{
    int e = blockIdx.x * 8 + (threadIdx.x >> 5);
    int lane = threadIdx.x & 31;
    if (e >= E) return;
    int s = src[e], d = dst[e];
    const float* hs = h + (size_t)s * 256;
    __builtin_prefetch(hs, 0, 1);                    // global_prefetch_b8
    float4 ex4 = *(const float4*)(ebuf + (size_t)e * 4);
    float4 dn4 = *(const float4*)(den + (size_t)d * 4);
    float alpha[4] = { ex4.x / (dn4.x + 1e-16f), ex4.y / (dn4.y + 1e-16f),
                       ex4.z / (dn4.z + 1e-16f), ex4.w / (dn4.w + 1e-16f) };
    float* ag = agg + (size_t)d * 256;
#pragma unroll
    for (int j = 0; j < 8; ++j) {
        int c = lane + 32 * j;                       // head = c>>6 = j>>1
        atomicAdd(&ag[c], alpha[j >> 1] * hs[c]);
    }
}

// wave per node: x = relu(LN(agg + gat_b) * gamma + beta)
__global__ __launch_bounds__(256)
void ln_relu(const float* __restrict__ agg, const float* __restrict__ gb,
             const float* __restrict__ gamma, const float* __restrict__ beta,
             float* __restrict__ xout, int n)
{
    int node = blockIdx.x * 8 + (threadIdx.x >> 5);
    int lane = threadIdx.x & 31;
    if (node >= n) return;
    const float* ar = agg + (size_t)node * 256;
    float v[8], s = 0.f, s2 = 0.f;
#pragma unroll
    for (int j = 0; j < 8; ++j) {
        int c = lane + 32 * j;
        v[j] = ar[c] + gb[c];
        s += v[j]; s2 += v[j] * v[j];
    }
    s  = warp_reduce_sum(s);
    s2 = warp_reduce_sum(s2);
    float mu   = s * (1.f / 256.f);
    float var  = s2 * (1.f / 256.f) - mu * mu;
    float rstd = rsqrtf(var + 1e-5f);
    float* xo = xout + (size_t)node * 256;
#pragma unroll
    for (int j = 0; j < 8; ++j) {
        int c = lane + 32 * j;
        float y = (v[j] - mu) * rstd * gamma[c] + beta[c];
        xo[c] = (y > 0.f) ? y : 0.f;
    }
}

// wave per b: out[b] = <u,v>; out[B..] = u flat; out[B+B*256..] = v flat
__global__ __launch_bounds__(256)
void finalize(const float* __restrict__ x, float* __restrict__ out, int B)
{
    int b = blockIdx.x * 8 + (threadIdx.x >> 5);
    int lane = threadIdx.x & 31;
    if (b >= B) return;
    const float* u = x + (size_t)b * 256;
    const float* v = x + (size_t)(B + b) * 256;
    float* uo = out + (size_t)B + (size_t)b * 256;
    float* vo = out + (size_t)B + (size_t)B * 256 + (size_t)b * 256;
    float dp = 0.f;
#pragma unroll
    for (int j = 0; j < 8; ++j) {
        int c = lane + 32 * j;
        float uu = u[c], vv = v[c];
        dp += uu * vv;
        uo[c] = uu; vo[c] = vv;
    }
    dp = warp_reduce_sum(dp);
    if (lane == 0) out[b] = dp;
}

// ---------------------------------------------------------------------------
extern "C" void kernel_launch(void* const* d_in, const int* in_sizes, int n_in,
                              void* d_out, int out_size, void* d_ws, size_t ws_size,
                              hipStream_t stream)
{
    (void)n_in; (void)out_size; (void)ws_size;
    const float* text  = (const float*)d_in[0];
    const float* img   = (const float*)d_in[1];
    const float* uet   = (const float*)d_in[2];
    const float* iet   = (const float*)d_in[3];
    const float* ut_W  = (const float*)d_in[4];
    const float* ut_b  = (const float*)d_in[5];
    const float* ui_W  = (const float*)d_in[6];
    const float* ui_b  = (const float*)d_in[7];
    const float* it_W  = (const float*)d_in[8];
    const float* it_b  = (const float*)d_in[9];
    const float* ii_W  = (const float*)d_in[10];
    const float* ii_b  = (const float*)d_in[11];
    const float* ua_Wv = (const float*)d_in[12];
    const float* ua_bv = (const float*)d_in[13];
    const float* ua_Wo = (const float*)d_in[14];
    const float* ua_bo = (const float*)d_in[15];
    const float* ia_Wv = (const float*)d_in[16];
    const float* ia_bv = (const float*)d_in[17];
    const float* ia_Wo = (const float*)d_in[18];
    const float* ia_bo = (const float*)d_in[19];
    const float* gW[3] = {(const float*)d_in[20], (const float*)d_in[21], (const float*)d_in[22]};
    const float* att_src = (const float*)d_in[23];   // [3][4][64] -> [3][256]
    const float* att_dst = (const float*)d_in[24];
    const float* gatB  = (const float*)d_in[25];     // [3][256]
    const float* resW  = (const float*)d_in[26];     // [2][256][256]
    const float* resB  = (const float*)d_in[27];     // [2][256]
    const float* lnG   = (const float*)d_in[28];
    const float* lnBt  = (const float*)d_in[29];
    const int*   uid   = (const int*)d_in[30];
    const int*   iid   = (const int*)d_in[31];
    const int*   eidx  = (const int*)d_in[32];

    const int B  = in_sizes[30];
    const int E  = in_sizes[32] / 2;
    const int Tn = in_sizes[0] / 1000;
    const int In = in_sizes[1] / 512;
    const int n  = 2 * B;
    const int* esrc = eidx;
    const int* edst = eidx + E;

    // ---- workspace layout ----
    float* ws = (float*)d_ws;
    size_t off = 0;
    float* xbig = ws + off; off += (size_t)n * 512;   // layer-0 input [2B,512]; A1 reused later
    float* regB = ws + off; off += (size_t)n * 256;   // um|im, later x between layers
    float* regC = ws + off; off += (size_t)n * 256;   // attn temp, later h
    float* regD = ws + off; off += (size_t)n * 256;   // GAT aggregation
    float* esb  = ws + off; off += (size_t)n * 4;
    float* edb  = ws + off; off += (size_t)n * 4;
    float* mbuf = ws + off; off += (size_t)n * 4;
    float* den  = ws + off; off += (size_t)n * 4;
    float* ebuf = ws + off; off += (size_t)E * 4;
    float* A1   = xbig;                               // [2B,256] reuse (layer-0 x dead)

    // ---- f16 weight pool (converted once; ~2 MB) ----
    const float* wsrc[13] = { ut_W, ui_W, it_W, ii_W, ua_Wv, ua_Wo, ia_Wv, ia_Wo,
                              gW[0], gW[1], gW[2], resW, resW + 256 * 256 };
    const int    wcnt[13] = { 1000 * 128, 512 * 128, 1000 * 128, 512 * 128,
                              256 * 256, 256 * 256, 256 * 256, 256 * 256,
                              512 * 256, 256 * 256, 256 * 256, 256 * 256, 256 * 256 };
    _Float16* wf16[13];
    {
        _Float16* wp = (_Float16*)(ws + off);
        for (int i = 0; i < 13; ++i) {
            wf16[i] = wp; wp += wcnt[i];
            cvt_f32_f16<<<(wcnt[i] + 255) / 256, 256, 0, stream>>>(wsrc[i], wf16[i], wcnt[i]);
        }
    }

    auto gemm = [&](const float* X, int ldX, const int* idx, int idxMod,
                    const _Float16* W16, int K, int N, const float* bias,
                    const float* resid, int ldR, float* Y, int ldY, int M) {
        dim3 g((M + 127) / 128, N / 128);
        gemm_wmma_f16<<<g, 256, 0, stream>>>(X, ldX, idx, idxMod, W16, K, N,
                                             bias, resid, ldR, Y, ldY, M);
    };

    // ---- Stage A: modality projections + attention MLPs + node assembly ----
    float* um = regB;                     // [B,256]
    float* im = regB + (size_t)B * 256;   // [B,256]
    gemm(text, 1000, uid, Tn, wf16[0], 1000, 128, ut_b, nullptr, 0, um,       256, B);
    gemm(img,  512,  uid, In, wf16[1], 512,  128, ui_b, nullptr, 0, um + 128, 256, B);
    gemm(text, 1000, iid, Tn, wf16[2], 1000, 128, it_b, nullptr, 0, im,       256, B);
    gemm(img,  512,  iid, In, wf16[3], 512,  128, ii_b, nullptr, 0, im + 128, 256, B);
    gemm(um,   256, nullptr, 0, wf16[4], 256, 256, ua_bv, nullptr, 0, regC, 256, B);
    gemm(regC, 256, nullptr, 0, wf16[5], 256, 256, ua_bo, nullptr, 0, xbig + 256, 512, B);
    gemm(im,   256, nullptr, 0, wf16[6], 256, 256, ia_bv, nullptr, 0, regC, 256, B);
    gemm(regC, 256, nullptr, 0, wf16[7], 256, 256, ia_bo, nullptr, 0,
         xbig + (size_t)B * 512 + 256, 512, B);
    {
        int tot = n * 64;
        embed_copy<<<(tot + 255) / 256, 256, 0, stream>>>(uet, iet, uid, iid, xbig, B);
    }

    // ---- Stage B: 3 GAT layers ----
    const float* xin = xbig; int ldx = 512;
    float* hbuf = regC;
    for (int L = 0; L < 3; ++L) {
        if (L > 0) {
            // xr = x + x @ res_W[L-1] + res_b[L-1]  -> A1
            gemm(xin, 256, nullptr, 0, wf16[11 + (L - 1)], 256, 256,
                 resB + (L - 1) * 256, xin, 256, A1, 256, n);
            xin = A1; ldx = 256;
        }
        gemm(xin, ldx, nullptr, 0, wf16[8 + L], ldx, 256, nullptr, nullptr, 0, hbuf, 256, n);
        gat_scores<<<(n + 7) / 8, 256, 0, stream>>>(hbuf, att_src + L * 256,
                                                    att_dst + L * 256, esb, edb, n);
        {
            size_t tot = (size_t)n * 256;
            init_layer<<<(int)((tot + 255) / 256), 256, 0, stream>>>(mbuf, den, regD, n);
        }
        { int tot = E * 4; edge_max<<<(tot + 255) / 256, 256, 0, stream>>>(esrc, edst, esb, edb, ebuf, mbuf, E); }
        { int tot = n * 4; fix_m<<<(tot + 255) / 256, 256, 0, stream>>>(mbuf, tot); }
        { int tot = E * 4; edge_exp<<<(tot + 255) / 256, 256, 0, stream>>>(edst, ebuf, mbuf, den, E); }
        edge_agg<<<(E + 7) / 8, 256, 0, stream>>>(esrc, edst, ebuf, den, hbuf, regD, E);
        ln_relu<<<(n + 7) / 8, 256, 0, stream>>>(regD, gatB + L * 256, lnG + L * 256,
                                                 lnBt + L * 256, regB, n);
        xin = regB; ldx = 256;
    }

    // ---- Stage C: outputs ----
    finalize<<<(B + 7) / 8, 256, 0, stream>>>(regB, (float*)d_out, B);
}
